// KoopmanOperator_84293028152117
// MI455X (gfx1250) — matrix operand order repaired and checked by reading, not compiled, not run on hardware
//
#include <hip/hip_runtime.h>
#include <hip/hip_bf16.h>

typedef __attribute__((ext_vector_type(16))) _Float16 v16h;
typedef __attribute__((ext_vector_type(8)))  _Float16 v8h;
typedef __attribute__((ext_vector_type(8)))  float    v8f;

#define DT_F    0.01f
#define NCOM    64
#define NREAL   64
#define KDIM    192     // 2*NCOM + NREAL
#define BATCH   4096
#define TSTEPS  64

// f16 weight copy layout inside d_ws (element offsets)
#define OFF_C1  0                    // cW1: 128 x 192
#define OFF_C2  24576                // cW2: 128 x 128
#define OFF_R1  40960                // rW1:  64 x 192
#define OFF_R2  53248                // rW2:  64 x  64
#define W_TOTAL 57344

// ---------------------------------------------------------------------------
// Fragment loaders for v_wmma_f32_16x16x32_f16 (wave32).
//
// A (16x32, f16), row m striped per lane:
//   lane<16 : m=lane,    elems 0..15 -> K=kt*32+0..15
//   lane>=16: m=lane-16, elems 0..15 -> K=kt*32+16..31
// => two contiguous v8h (16B) loads per lane.
__device__ inline v16h load_a_frag(const _Float16* base, int Kdim, int kt, int lane) {
    const int hi = (lane >> 4) & 1;
    const int m  = lane & 15;
    const _Float16* p = base + m * Kdim + kt * 32 + hi * 16;
    union { v16h v; v8h h[2]; } f;
    f.h[0] = *(const v8h*)(p);
    f.h[1] = *(const v8h*)(p + 8);
    return f.v;
}

// B (32x16, f16) for y @ W^T: B[k][n] = W[n][k], W row-major [Nout][Kin].
//   lane<16 : n=lane,    elems 0..15 -> K=kt*32+0..15
//   lane>=16: n=lane-16, elems 0..15 -> K=kt*32+16..31
// => one contiguous 32-B run per lane from W's row n.
__device__ inline v16h load_b_frag(const _Float16* W, int Kdim, int kt, int nt, int lane) {
    const int hi = (lane >> 4) & 1;
    const int n  = nt * 16 + (lane & 15);
    const _Float16* p = W + n * Kdim + kt * 32 + hi * 16;
    return *(const v16h*)(p);
}

// ---------------------------------------------------------------------------
// Prologue: convert fp32 weights to a packed f16 copy in workspace.
__global__ void convert_weights_kernel(const float* __restrict__ cW1,
                                       const float* __restrict__ cW2,
                                       const float* __restrict__ rW1,
                                       const float* __restrict__ rW2,
                                       _Float16* __restrict__ ws) {
    int i = blockIdx.x * blockDim.x + threadIdx.x;
    if (i >= W_TOTAL) return;
    float v;
    if (i < OFF_C2)      v = cW1[i - OFF_C1];
    else if (i < OFF_R1) v = cW2[i - OFF_C2];
    else if (i < OFF_R2) v = rW1[i - OFF_R1];
    else                 v = rW2[i - OFF_R2];
    ws[i] = (_Float16)v;
}

// ---------------------------------------------------------------------------
// Main rollout: one block = 4 waves = one 16-row batch tile, loops over T.
__global__ __launch_bounds__(128, 2)
void koopman_rollout_kernel(const float* __restrict__ x,
                            const float* __restrict__ cb1, const float* __restrict__ cb2,
                            const float* __restrict__ rb1, const float* __restrict__ rb2,
                            const _Float16* __restrict__ wf16,
                            float* __restrict__ out) {
    __shared__ __attribute__((aligned(16))) float    yl[16 * KDIM];   // fp32 state
    __shared__ __attribute__((aligned(16))) _Float16 y16[16 * KDIM];  // f16 snapshot (A of layer 1)
    __shared__ __attribute__((aligned(16))) _Float16 h1c[16 * 128];   // relu hidden, complex branch
    __shared__ __attribute__((aligned(16))) _Float16 h1r[16 * 64];    // relu hidden, real branch
    __shared__ __attribute__((aligned(16))) float    hbuf[16 * 128];  // mu/omega interleaved
    __shared__ __attribute__((aligned(16))) float    rbuf[16 * 64];   // real rates

    const int tid  = threadIdx.x;
    const int wave = tid >> 5;
    const int lane = tid & 31;
    const int b0   = blockIdx.x * 16;

    const _Float16* wc1 = wf16 + OFF_C1;
    const _Float16* wc2 = wf16 + OFF_C2;
    const _Float16* wr1 = wf16 + OFF_R1;
    const _Float16* wr2 = wf16 + OFF_R2;

    // y = x[:, 0, :]
    for (int i = tid; i < 16 * KDIM; i += 128) {
        int m = i / KDIM, k = i % KDIM;
        yl[i] = x[((size_t)(b0 + m) * TSTEPS) * KDIM + k];
    }
    __syncthreads();

    for (int t = 0; t < TSTEPS; ++t) {
        // f16 snapshot of state for the layer-1 GEMMs
        for (int i = tid; i < 16 * KDIM; i += 128) y16[i] = (_Float16)yl[i];
        __syncthreads();

        // ---- layer 1 (both branches): h = relu(y @ W1^T + b1) ----
        // 12 N-tile jobs (8 complex + 4 real) split across the 4 waves.
        for (int job = wave; job < 12; job += 4) {
            const bool      isC  = job < 8;
            const int       nt   = isC ? job : job - 8;
            const _Float16* W    = isC ? wc1 : wr1;
            const float*    bs   = isC ? cb1 : rb1;
            _Float16*       dst  = isC ? h1c : h1r;
            const int       Nout = isC ? 128 : 64;
            const float bv = bs[nt * 16 + (lane & 15)];
            v8f acc = {bv, bv, bv, bv, bv, bv, bv, bv};
            for (int kt = 0; kt < 6; ++kt) {   // K = 192 = 6 * 32
                v16h a = load_a_frag(y16, KDIM, kt, lane);
                v16h b = load_b_frag(W, KDIM, kt, nt, lane);
                acc = __builtin_amdgcn_wmma_f32_16x16x32_f16(false, a, false, b,
                                                             (short)0, acc, false, false);
            }
            const int hi = (lane >> 4) & 1;
            const int n  = nt * 16 + (lane & 15);
#pragma unroll
            for (int r = 0; r < 8; ++r) {
                float v = acc[r];
                v = v > 0.0f ? v : 0.0f;                 // ReLU
                dst[(r + hi * 8) * Nout + n] = (_Float16)v;
            }
        }
        __syncthreads();

        // ---- layer 2 (both branches): h2 = h @ W2^T + b2 ----
        for (int job = wave; job < 12; job += 4) {
            const bool      isC  = job < 8;
            const int       nt   = isC ? job : job - 8;
            const _Float16* W    = isC ? wc2 : wr2;
            const float*    bs   = isC ? cb2 : rb2;
            const _Float16* src  = isC ? h1c : h1r;
            float*          dst2 = isC ? hbuf : rbuf;
            const int       Kd   = isC ? 128 : 64;
            const int       Nout = isC ? 128 : 64;
            const int       nk   = isC ? 4 : 2;
            const float bv = bs[nt * 16 + (lane & 15)];
            v8f acc = {bv, bv, bv, bv, bv, bv, bv, bv};
            for (int kt = 0; kt < nk; ++kt) {
                v16h a = load_a_frag(src, Kd, kt, lane);
                v16h b = load_b_frag(W, Kd, kt, nt, lane);
                acc = __builtin_amdgcn_wmma_f32_16x16x32_f16(false, a, false, b,
                                                             (short)0, acc, false, false);
            }
            const int hi = (lane >> 4) & 1;
            const int n  = nt * 16 + (lane & 15);
#pragma unroll
            for (int r = 0; r < 8; ++r) dst2[(r + hi * 8) * Nout + n] = acc[r];
        }
        __syncthreads();

        // ---- elementwise Koopman update (in place on yl) ----
        for (int i = tid; i < 16 * NCOM; i += 128) {
            int m = i >> 6, j = i & 63;
            float mu = hbuf[m * 128 + 2 * j];
            float om = hbuf[m * 128 + 2 * j + 1];
            float e  = __expf(DT_F * mu);
            float sn, cs;
            __sincosf(DT_F * om, &sn, &cs);
            float ye = yl[m * KDIM + 2 * j];
            float yo = yl[m * KDIM + 2 * j + 1];
            yl[m * KDIM + 2 * j]     = e * (cs * ye - sn * yo);
            yl[m * KDIM + 2 * j + 1] = e * (sn * ye + cs * yo);
        }
        for (int i = tid; i < 16 * NREAL; i += 128) {
            int m = i >> 6, n = i & 63;
            yl[m * KDIM + 2 * NCOM + n] *= __expf(DT_F * rbuf[m * 64 + n]);
        }
        __syncthreads();

        // emit y_new -> out[b, t, :]
        for (int i = tid; i < 16 * KDIM; i += 128) {
            int m = i / KDIM, k = i % KDIM;
            out[(((size_t)(b0 + m)) * TSTEPS + t) * KDIM + k] = yl[i];
        }
        __syncthreads();
    }
}

// ---------------------------------------------------------------------------
extern "C" void kernel_launch(void* const* d_in, const int* in_sizes, int n_in,
                              void* d_out, int out_size, void* d_ws, size_t ws_size,
                              hipStream_t stream) {
    (void)in_sizes; (void)n_in; (void)out_size; (void)ws_size;
    const float* x   = (const float*)d_in[0];
    const float* cW1 = (const float*)d_in[1];
    const float* cb1 = (const float*)d_in[2];
    const float* cW2 = (const float*)d_in[3];
    const float* cb2 = (const float*)d_in[4];
    const float* rW1 = (const float*)d_in[5];
    const float* rb1 = (const float*)d_in[6];
    const float* rW2 = (const float*)d_in[7];
    const float* rb2 = (const float*)d_in[8];
    _Float16* wf16 = (_Float16*)d_ws;
    float*    out  = (float*)d_out;

    convert_weights_kernel<<<(W_TOTAL + 255) / 256, 256, 0, stream>>>(cW1, cW2, rW1, rW2, wf16);
    koopman_rollout_kernel<<<BATCH / 16, 128, 0, stream>>>(x, cb1, cb2, rb1, rb2, wf16, out);
}
